// MultiHeadAttention3D_54013508714616
// MI455X (gfx1250) — matrix operand order repaired and checked by reading, not compiled, not run on hardware
//
#include <hip/hip_runtime.h>

typedef __attribute__((ext_vector_type(16))) _Float16 v16h;
typedef __attribute__((ext_vector_type(8)))  float    v8f;

#define EMBED 768
#define NHEAD 12
#define HDIM  64
#define BB    2
#define NSEQ  2048
#define C3    (3*EMBED)

__device__ __forceinline__ v8f wmma16(v16h a, v16h b, v8f c) {
  return __builtin_amdgcn_wmma_f32_16x16x32_f16(false, a, false, b, (short)0, c,
                                                false, false);
}

// A-operand (16x32 f16). p = row base + k0 (row chosen per-lane: m = tile + lane%16)
// lane l: halves 0..7  -> K = (l/16)*8 + i ; halves 8..15 -> K = 16 + (l/16)*8 + i
__device__ __forceinline__ v16h load_a32(const float* p, int lane) {
  v16h a; const float* q = p + ((lane >> 4) << 3);
  #pragma unroll
  for (int i = 0; i < 8; i++) { a[i] = (_Float16)q[i]; a[8 + i] = (_Float16)q[16 + i]; }
  return a;
}
__device__ __forceinline__ v16h load_a16(const _Float16* p, int lane) {
  v16h a; const _Float16* q = p + ((lane >> 4) << 3);
  #pragma unroll
  for (int i = 0; i < 8; i++) { a[i] = q[i]; a[8 + i] = q[16 + i]; }
  return a;
}
// B-operand (32x16). p = row base + k0 (row = output col n = tile + lane%16)
// lane l: halves 0..7  -> K = (l/16)*8 + i ; halves 8..15 -> K = 16 + (l/16)*8 + i
__device__ __forceinline__ v16h load_b32(const float* p, int lane) {
  v16h b; const float* q = p + ((lane >> 4) << 3);
  #pragma unroll
  for (int i = 0; i < 8; i++) { b[i] = (_Float16)q[i]; b[8 + i] = (_Float16)q[16 + i]; }
  return b;
}
__device__ __forceinline__ v16h load_b16lds(const _Float16* p, int lane) {
  v16h b; const _Float16* q = p + ((lane >> 4) << 3);
  #pragma unroll
  for (int i = 0; i < 8; i++) { b[i] = q[i]; b[8 + i] = q[16 + i]; }
  return b;
}

// async memory->LDS copy of one 16B chunk per lane (tracked by ASYNCcnt)
__device__ __forceinline__ void async_b128(const _Float16* gsrc, _Float16* lds_dst) {
  const unsigned long long ga = (unsigned long long)gsrc;
  const unsigned la = (unsigned)(unsigned long long)lds_dst;  // low 32 bits = LDS offset
  asm volatile("global_load_async_to_lds_b128 %0, %1, off"
               :: "v"(la), "v"(ga) : "memory");
}
__device__ __forceinline__ void wait_async0() {
  asm volatile("s_wait_asynccnt 0x0" ::: "memory");
}

// ---------------- Kernel 1: fused QKV projection (NT GEMM, K=768) ----------------
// one wave = 16x64 output tile (A reused across 4 WMMAs per K-step).
// Writes Q,K as [b][h][n][d] f16, V transposed [b][h][d][n] f16.
__global__ __launch_bounds__(128) void k_qkv(const float* __restrict__ x,
                                             const float* __restrict__ w,
                                             const float* __restrict__ bias,
                                             _Float16* __restrict__ qf,
                                             _Float16* __restrict__ kf,
                                             _Float16* __restrict__ vt) {
  const int lane = threadIdx.x & 31;
  const int wave = (blockIdx.x * blockDim.x + threadIdx.x) >> 5;
  const int NT = C3 / 64;                       // 36 column tiles
  const int mt = wave / NT, nt = wave % NT;
  const int lo = lane & 15, hi = lane >> 4;
  const int am = mt * 16 + lo;                  // row of X  (b*N+n)
  const int bcol = nt * 64;                     // tile base column (64-aligned)
  const float* xr = x + (size_t)am * EMBED;
  const float* wr = w + (size_t)(bcol + lo) * EMBED;
  v8f acc[4];
  #pragma unroll
  for (int s = 0; s < 4; s++) acc[s] = (v8f){0.f,0.f,0.f,0.f,0.f,0.f,0.f,0.f};
  for (int k0 = 0; k0 < EMBED; k0 += 32) {
    v16h a = load_a32(xr + k0, lane);
    #pragma unroll
    for (int s = 0; s < 4; s++) {
      v16h b = load_b32(wr + (size_t)s * 16 * EMBED + k0, lane);
      acc[s] = wmma16(a, b, acc[s]);
    }
  }
  const int which = bcol / EMBED;               // 0=q, 1=k, 2=v (tile-uniform)
  const int h = (bcol % EMBED) / HDIM;
  #pragma unroll
  for (int s = 0; s < 4; s++) {
    const int dd = s * 16 + lo;                 // dim within head, 0..63
    const float bv = bias[bcol + dd];
    #pragma unroll
    for (int r = 0; r < 8; r++) {
      const int m = mt * 16 + hi * 8 + r;       // global row in [0, B*N)
      const int bidx = m / NSEQ, n = m % NSEQ;
      const _Float16 hval = (_Float16)(acc[s][r] + bv);
      const size_t bh = (size_t)bidx * NHEAD + h;
      if (which == 0)      qf[(bh * NSEQ + n) * HDIM + dd] = hval;
      else if (which == 1) kf[(bh * NSEQ + n) * HDIM + dd] = hval;
      else                 vt[(bh * HDIM + dd) * NSEQ + n] = hval;
    }
  }
}

// ---------------- Kernel 2: fused flash attention ----------------
// one block = 4 waves = 64 query rows of one (b,h); K/V chunks async-staged in
// LDS once per block (shared by all 4 waves); online softmax over 32-key chunks.
#define WPB 4
__global__ __launch_bounds__(32*WPB) void k_attn(const float* __restrict__ rel_bias,
                                                 const int* __restrict__ mask,
                                                 const _Float16* __restrict__ qf,
                                                 const _Float16* __restrict__ kf,
                                                 const _Float16* __restrict__ vt,
                                                 _Float16* __restrict__ of) {
  __shared__ _Float16 pst[WPB][16 * 32];        // P staging (C-layout -> A-layout)
  __shared__ _Float16 Ksm[32 * HDIM];           // K chunk, [key][d]
  __shared__ _Float16 Vsm[HDIM * 32];           // V chunk, [d][key]
  const int t = threadIdx.x;
  const int lane = t & 31;
  const int wslot = t >> 5;
  const int QG = NSEQ / 64;                     // 32 query groups per (b,h)
  const int bid = blockIdx.x;
  const int b  = bid / (NHEAD * QG);
  const int rm = bid % (NHEAD * QG);
  const int h  = rm / QG, qg = rm % QG;
  const int qbase = (qg * 4 + wslot) * 16;      // this wave's 16 query rows
  const int lo = lane & 15, hi = lane >> 4;

  const size_t bh = (size_t)b * NHEAD + h;
  const _Float16* qrow  = qf + bh * NSEQ * HDIM;
  const _Float16* krow  = kf + bh * NSEQ * HDIM;   // [key][d], chunk is contiguous
  const _Float16* vbase = vt + bh * HDIM * NSEQ;   // [d][key]
  const float* rb   = rel_bias + (size_t)h * NSEQ * NSEQ;
  const int*   mrow = mask + (size_t)b * NSEQ * NSEQ;

  // Q A-operands (K = d = 64 -> two 16x32 operands), loaded once
  v16h A0, A1;
  {
    const _Float16* p = qrow + (size_t)(qbase + lo) * HDIM + (hi << 3);
    #pragma unroll
    for (int i = 0; i < 8; i++) {
      A0[i] = p[i];      A0[8 + i] = p[16 + i];
      A1[i] = p[32 + i]; A1[8 + i] = p[48 + i];
    }
  }
  float rmax[8], rsum[8];
  #pragma unroll
  for (int r = 0; r < 8; r++) { rmax[r] = -1e30f; rsum[r] = 0.f; }
  v8f oacc[4];
  #pragma unroll
  for (int nb = 0; nb < 4; nb++) oacc[nb] = (v8f){0.f,0.f,0.f,0.f,0.f,0.f,0.f,0.f};
  const float scale = 0.125f;                   // 1/sqrt(64)
  const int me_base = qbase + hi * 8;

  for (int j0 = 0; j0 < NSEQ; j0 += 32) {
    // ---- async stage K chunk (32x64, contiguous 4KB) and V chunk (64x32) ----
    {
      const _Float16* ks = krow + (size_t)j0 * HDIM;     // 2048 halves contiguous
      #pragma unroll
      for (int it = 0; it < 2; it++) {
        const int chunk = t + it * 128;                  // 256 x 16B chunks
        async_b128(ks + chunk * 8, &Ksm[chunk * 8]);
      }
      #pragma unroll
      for (int it = 0; it < 2; it++) {
        const int chunk = t + it * 128;                  // 64 rows x 4 chunks
        const int seg = chunk >> 2, sofs = (chunk & 3) * 8;
        async_b128(vbase + (size_t)seg * NSEQ + j0 + sofs, &Vsm[seg * 32 + sofs]);
      }
      wait_async0();
      __syncthreads();
    }
    // ---- scores: two 16-col tiles, K=64 over d, B-operands from LDS ----
    v8f s0 = {0.f,0.f,0.f,0.f,0.f,0.f,0.f,0.f};
    v8f s1 = s0;
    {
      v16h B0 = load_b16lds(&Ksm[(lo)      * HDIM],      lane);
      v16h B1 = load_b16lds(&Ksm[(lo)      * HDIM + 32], lane);
      v16h B2 = load_b16lds(&Ksm[(16 + lo) * HDIM],      lane);
      v16h B3 = load_b16lds(&Ksm[(16 + lo) * HDIM + 32], lane);
      s0 = wmma16(A0, B0, s0); s0 = wmma16(A1, B1, s0);
      s1 = wmma16(A0, B2, s1); s1 = wmma16(A1, B3, s1);
    }
    // ---- bias + mask + online softmax ----
    #pragma unroll
    for (int r = 0; r < 8; r++) {
      const int me = me_base + r;
      const int n0 = j0 + lo, n1 = j0 + 16 + lo;
      // prefetch next chunk of the streaming rel_bias row (global_prefetch_b8)
      if (j0 + 32 < NSEQ) __builtin_prefetch(rb + (size_t)me * NSEQ + n0 + 32, 0, 1);
      float v0 = s0[r] * scale + rb[(size_t)me * NSEQ + n0];
      float v1 = s1[r] * scale + rb[(size_t)me * NSEQ + n1];
      if (mrow[(size_t)me * NSEQ + n0] == 0) v0 = -1e30f;
      if (mrow[(size_t)me * NSEQ + n1] == 0) v1 = -1e30f;

      float cm = fmaxf(v0, v1);                 // reduce over 16 lanes of the half
      #pragma unroll
      for (int off = 1; off < 16; off <<= 1) cm = fmaxf(cm, __shfl_xor(cm, off, 32));
      const float mnew = fmaxf(rmax[r], cm);
      const float corr = __expf(rmax[r] - mnew);
      const float p0 = (v0 <= -1e29f) ? 0.f : __expf(v0 - mnew);
      const float p1 = (v1 <= -1e29f) ? 0.f : __expf(v1 - mnew);
      float ps = p0 + p1;
      #pragma unroll
      for (int off = 1; off < 16; off <<= 1) ps += __shfl_xor(ps, off, 32);
      rsum[r] = rsum[r] * corr + ps;
      rmax[r] = mnew;
      oacc[0][r] *= corr; oacc[1][r] *= corr; oacc[2][r] *= corr; oacc[3][r] *= corr;
      const int mr = hi * 8 + r;
      pst[wslot][mr * 32 + lo]      = (_Float16)p0;
      pst[wslot][mr * 32 + 16 + lo] = (_Float16)p1;
    }
    __syncthreads();
    // ---- P as A-operand (16x32) from LDS ----
    v16h AP;
    {
      const _Float16* p = &pst[wslot][lo * 32 + (hi << 3)];
      #pragma unroll
      for (int i = 0; i < 8; i++) { AP[i] = p[i]; AP[8 + i] = p[16 + i]; }
    }
    // ---- P @ V : 4 col-blocks of d, B-operands from LDS ----
    #pragma unroll
    for (int nb = 0; nb < 4; nb++) {
      v16h BV = load_b16lds(&Vsm[(nb * 16 + lo) * 32], lane);
      oacc[nb] = wmma16(AP, BV, oacc[nb]);
    }
    __syncthreads();   // all waves done with Ksm/Vsm/pst before next stage
  }
  // ---- normalize + store O as f16 [b][n][h*d + c] ----
  #pragma unroll
  for (int nb = 0; nb < 4; nb++) {
    #pragma unroll
    for (int r = 0; r < 8; r++) {
      const int me = me_base + r;
      const float val = oacc[nb][r] / rsum[r];
      of[((size_t)b * NSEQ + me) * EMBED + h * HDIM + nb * 16 + lo] = (_Float16)val;
    }
  }
}

// ---------------- Kernel 3: output projection (NT GEMM, K=768, f32 out) ----------------
// one wave = 16x64 output tile.
__global__ __launch_bounds__(128) void k_proj(const _Float16* __restrict__ of,
                                              const float* __restrict__ pw,
                                              const float* __restrict__ pb,
                                              float* __restrict__ out) {
  const int lane = threadIdx.x & 31;
  const int wave = (blockIdx.x * blockDim.x + threadIdx.x) >> 5;
  const int NT = EMBED / 64;                    // 12 column tiles
  const int mt = wave / NT, nt = wave % NT;
  const int lo = lane & 15, hi = lane >> 4;
  const int am = mt * 16 + lo;
  const int bcol = nt * 64;
  const _Float16* ar = of + (size_t)am * EMBED;
  const float*    wr = pw + (size_t)(bcol + lo) * EMBED;
  v8f acc[4];
  #pragma unroll
  for (int s = 0; s < 4; s++) acc[s] = (v8f){0.f,0.f,0.f,0.f,0.f,0.f,0.f,0.f};
  for (int k0 = 0; k0 < EMBED; k0 += 32) {
    v16h a = load_a16(ar + k0, lane);
    #pragma unroll
    for (int s = 0; s < 4; s++) {
      v16h b = load_b32(wr + (size_t)s * 16 * EMBED + k0, lane);
      acc[s] = wmma16(a, b, acc[s]);
    }
  }
  #pragma unroll
  for (int s = 0; s < 4; s++) {
    const int n = bcol + s * 16 + lo;
    const float bv = pb[n];
    #pragma unroll
    for (int r = 0; r < 8; r++) {
      const int m = mt * 16 + hi * 8 + r;
      out[(size_t)m * EMBED + n] = acc[s][r] + bv;
    }
  }
}

extern "C" void kernel_launch(void* const* d_in, const int* in_sizes, int n_in,
                              void* d_out, int out_size, void* d_ws, size_t ws_size,
                              hipStream_t stream) {
  const float* x        = (const float*)d_in[0];
  const int*   mask     = (const int*)d_in[1];
  const float* qkv_w    = (const float*)d_in[2];
  const float* qkv_b    = (const float*)d_in[3];
  const float* rel_bias = (const float*)d_in[4];
  const float* proj_w   = (const float*)d_in[5];
  const float* proj_b   = (const float*)d_in[6];
  float* out = (float*)d_out;

  const size_t qkv_elems = (size_t)BB * NHEAD * NSEQ * HDIM;  // 3,145,728 halves each
  _Float16* qf = (_Float16*)d_ws;
  _Float16* kf = qf + qkv_elems;
  _Float16* vt = kf + qkv_elems;
  _Float16* of = vt + qkv_elems;   // [B][N][EMBED] f16; total ws ~25.2 MB

  {
    const int waves = (BB * NSEQ / 16) * (C3 / 64);           // 9216
    k_qkv<<<dim3(waves / 4), dim3(128), 0, stream>>>(x, qkv_w, qkv_b, qf, kf, vt);
  }
  {
    const int blocks = BB * NHEAD * (NSEQ / 64);              // 768
    k_attn<<<dim3(blocks), dim3(32 * WPB), 0, stream>>>(rel_bias, mask, qf, kf, vt, of);
  }
  {
    const int waves = (BB * NSEQ / 16) * (EMBED / 64);        // 3072
    k_proj<<<dim3(waves / 4), dim3(128), 0, stream>>>(of, proj_w, proj_b, out);
  }
}
